// LiquidNeuralNetwork_75866302316812
// MI455X (gfx1250) — compile-verified, matrix operand order checked by reading, not run on hardware
//
#include <hip/hip_runtime.h>

typedef __attribute__((ext_vector_type(16))) __bf16 v16bf;
typedef __attribute__((ext_vector_type(8)))  float  v8f;

union Frag {
    uint4  u[2];
    v16bf  v;
};

__device__ inline unsigned short f2bf(float f) {
    union { float f; unsigned u; } a; a.f = f;
    unsigned r = a.u + 0x7FFFu + ((a.u >> 16) & 1u);   // round-to-nearest-even
    return (unsigned short)(r >> 16);
}

// ---------------------------------------------------------------------------
// f32 -> bf16 bulk convert (weights)
// ---------------------------------------------------------------------------
__global__ void k_f32_to_bf16(const float* __restrict__ src,
                              unsigned short* __restrict__ dst, long n) {
    long i = (long)blockIdx.x * blockDim.x + threadIdx.x;
    long stride = (long)gridDim.x * blockDim.x;
    for (; i < n; i += stride) dst[i] = f2bf(src[i]);
}

// ---------------------------------------------------------------------------
// Embedding lookup + sinusoidal positional encoding -> bf16 activations
// ---------------------------------------------------------------------------
__global__ void k_embed_pe(const int* __restrict__ x,
                           const float* __restrict__ emb,
                           unsigned short* __restrict__ hb,
                           int S, int E) {
    long idx = (long)blockIdx.x * blockDim.x + threadIdx.x;  // over B*S*E
    int e = (int)(idx % E);
    long t = idx / E;            // row in [0, B*S)
    int s = (int)(t % S);
    int tok = x[t];
    float v = emb[(long)tok * E + e];
    int i2 = e & ~1;             // 2*i for the pair
    float div = __expf(-logf(10000.0f) * (float)i2 / (float)E);
    float ang = (float)s * div;
    v += (e & 1) ? __cosf(ang) : __sinf(ang);
    hb[idx] = f2bf(v);
}

// ---------------------------------------------------------------------------
// Generic bf16 WMMA GEMM:  C[m,n] = sum_k A[m,k] * B[n,k]  (+ bias[n])
// A: [M,K] K-major (lda), B: [N,K] K-major (ldb)  -- i.e. C = A * B^T
// Batched over blockIdx.z with offset = (z/zdiv)*s?hi + (z%zdiv)*s?lo.
// Block = 256 threads = 8 waves (2 waves/SIMD32), 128x128 output tile.
// K staged 64 at a time through LDS with register double-buffering.
// __launch_bounds__(256, 2) grants the VGPR budget for the staged slab +
// 64 accumulator VGPRs + fragments without scratch spills.
// ---------------------------------------------------------------------------
#define BM 128
#define BN 128
#define BK 64
#define LDS_STRIDE 72   // halves; 144B rows keep 16B alignment, stagger banks

template <bool ALIGNED>
__global__ void __launch_bounds__(256, 2)
k_gemm_bf16(const unsigned short* __restrict__ A, int lda,
            long sAhi, long sAlo,
            const unsigned short* __restrict__ B, int ldb,
            long sBhi, long sBlo,
            const float* __restrict__ bias,
            float* __restrict__ C,
            unsigned short* __restrict__ Cb, int ldc,
            long sChi, long sClo,
            int M, int N, int K, int zdiv) {
    __shared__ unsigned short As[BM * LDS_STRIDE];
    __shared__ unsigned short Bs[BN * LDS_STRIDE];

    const int tid  = threadIdx.x;
    const int lane = tid & 31;
    const int w    = tid >> 5;
    const int m0   = blockIdx.y * BM;
    const int n0   = blockIdx.x * BN;
    const int bz   = blockIdx.z;

    const long offA = (long)(bz / zdiv) * sAhi + (long)(bz % zdiv) * sAlo;
    const long offB = (long)(bz / zdiv) * sBhi + (long)(bz % zdiv) * sBlo;
    const long offC = (long)(bz / zdiv) * sChi + (long)(bz % zdiv) * sClo;

    const unsigned short* Ag = A + offA;
    const unsigned short* Bg = B + offB;

    // ---- per-thread staging assignment: 4 chunks of 16B for A and for B ----
    // chunk id c = tid + i*256 -> row = c>>3 (0..127), ch = c&7 (8 chunks/row)
    const unsigned short* pA[4];
    const unsigned short* pB[4];
    int  ldsOff[4];
    bool okA[4], okB[4];
#pragma unroll
    for (int i = 0; i < 4; ++i) {
        int c = tid + i * 256;
        int row = c >> 3, ch = c & 7;
        okA[i] = ALIGNED || (m0 + row < M);
        okB[i] = ALIGNED || (n0 + row < N);
        pA[i] = Ag + (long)(m0 + (okA[i] ? row : 0)) * lda + ch * 8;
        pB[i] = Bg + (long)(n0 + (okB[i] ? row : 0)) * ldb + ch * 8;
        ldsOff[i] = row * LDS_STRIDE + ch * 8;
    }

    // wave sub-tile: 4 waves stacked on M (32 rows each), 2 on N (64 cols each)
    const int wr = (w & 3) * 32;
    const int wc = (w >> 2) * 64;
    const int r  = lane & 15;
    const int hf = lane >> 4;

    v8f acc[2][4] = {};
    uint4 ra[4], rb[4];

    // prologue: fetch first K-slab into registers
#pragma unroll
    for (int i = 0; i < 4; ++i) {
        ra[i] = okA[i] ? *(const uint4*)(pA[i]) : uint4{0, 0, 0, 0};
        rb[i] = okB[i] ? *(const uint4*)(pB[i]) : uint4{0, 0, 0, 0};
    }

    for (int kk = 0; kk < K; kk += BK) {
        __syncthreads();   // previous compute finished reading LDS
#pragma unroll
        for (int i = 0; i < 4; ++i) {
            *(uint4*)(&As[ldsOff[i]]) = ra[i];
            *(uint4*)(&Bs[ldsOff[i]]) = rb[i];
        }
        __syncthreads();

        // overlap: fetch next slab into registers while WMMAs consume LDS
        if (kk + BK < K) {
#pragma unroll
            for (int i = 0; i < 4; ++i) {
                ra[i] = okA[i] ? *(const uint4*)(pA[i] + kk + BK) : uint4{0, 0, 0, 0};
                rb[i] = okB[i] ? *(const uint4*)(pB[i] + kk + BK) : uint4{0, 0, 0, 0};
            }
            if (kk + 2 * BK < K) {
                __builtin_prefetch((const void*)(pA[0] + kk + 2 * BK), 0, 1);
                __builtin_prefetch((const void*)(pB[0] + kk + 2 * BK), 0, 1);
            }
        }

        // ---- two 32-deep WMMA steps per slab ----
#pragma unroll
        for (int ks = 0; ks < BK; ks += 32) {
            Frag fa[2], fb[4];
#pragma unroll
            for (int i = 0; i < 2; ++i) {
                const unsigned short* p = &As[(wr + i * 16 + r) * LDS_STRIDE + ks];
                fa[i].u[0] = *(const uint4*)(p + hf * 8);
                fa[i].u[1] = *(const uint4*)(p + 16 + hf * 8);
            }
#pragma unroll
            for (int j = 0; j < 4; ++j) {
                const unsigned short* p = &Bs[(wc + j * 16 + r) * LDS_STRIDE + ks];
                fb[j].u[0] = *(const uint4*)(p + hf * 8);
                fb[j].u[1] = *(const uint4*)(p + 16 + hf * 8);
            }
#pragma unroll
            for (int i = 0; i < 2; ++i)
#pragma unroll
                for (int j = 0; j < 4; ++j)
                    acc[i][j] = __builtin_amdgcn_wmma_f32_16x16x32_bf16(
                        false, fa[i].v, false, fb[j].v, (short)0, acc[i][j],
                        false, false);
        }
    }

    // ---- epilogue: C layout (VGPR rr: rows rr / rr+8, lanes = cols) ----
#pragma unroll
    for (int i = 0; i < 2; ++i)
#pragma unroll
        for (int j = 0; j < 4; ++j)
#pragma unroll
            for (int rr = 0; rr < 8; ++rr) {
                int row = m0 + wr + i * 16 + hf * 8 + rr;
                int col = n0 + wc + j * 16 + r;
                if (ALIGNED || (row < M && col < N)) {
                    float v = acc[i][j][rr];
                    if (bias) v += bias[col];
                    long o = offC + (long)row * ldc + col;
                    if (C)  C[o]  = v;
                    if (Cb) Cb[o] = f2bf(v);
                }
            }
}

// ---------------------------------------------------------------------------
// Liquid ODE scan (sequential over S; one thread per (batch, channel))
//   u' = u + dt*(-u + I)/tau ; s' = s + dt*(-s + u_old)/tau
// ---------------------------------------------------------------------------
__global__ void k_liquid_scan(const float* __restrict__ I,
                              const float* __restrict__ tau,
                              const float* __restrict__ log_dt,
                              float* __restrict__ Sout,
                              int Bn, int S, int O) {
    int t = blockIdx.x * blockDim.x + threadIdx.x;
    if (t >= Bn * O) return;
    int b = t / O, o = t % O;
    float dt = __expf(log_dt[0]);
    float itau = dt / tau[o];
    float u = 0.0f, s = 0.0f;
    const float* Ip = I + (long)b * S * O + o;
    float* Sp = Sout + (long)b * S * O + o;
    for (int st = 0; st < S; ++st) {
        float It = Ip[(long)st * O];
        float un = u + itau * (It - u);
        float sn = s + itau * (u - s);   // uses OLD u, as in reference
        u = un; s = sn;
        Sp[(long)st * O] = sn;
    }
}

// ---------------------------------------------------------------------------
// Fused tanh + LayerNorm -> bf16.  One block per row, D = 512, 256 threads.
// ---------------------------------------------------------------------------
__global__ void k_tanh_ln(const float* __restrict__ X,
                          const float* __restrict__ g,
                          const float* __restrict__ be,
                          unsigned short* __restrict__ out, int D) {
    __shared__ float s1[256], s2[256];
    long row = blockIdx.x;
    const float* xp = X + row * D;
    float v0 = tanhf(xp[threadIdx.x]);
    float v1 = tanhf(xp[threadIdx.x + 256]);
    s1[threadIdx.x] = v0 + v1;
    s2[threadIdx.x] = v0 * v0 + v1 * v1;
    __syncthreads();
    for (int wd = 128; wd > 0; wd >>= 1) {
        if (threadIdx.x < wd) {
            s1[threadIdx.x] += s1[threadIdx.x + wd];
            s2[threadIdx.x] += s2[threadIdx.x + wd];
        }
        __syncthreads();
    }
    float mean = s1[0] / (float)D;
    float var  = s2[0] / (float)D - mean * mean;
    float rstd = rsqrtf(var + 1e-5f);
    int c0 = threadIdx.x, c1 = threadIdx.x + 256;
    out[row * D + c0] = f2bf((v0 - mean) * rstd * g[c0] + be[c0]);
    out[row * D + c1] = f2bf((v1 - mean) * rstd * g[c1] + be[c1]);
}

// ---------------------------------------------------------------------------
// Row softmax (N = 1024) with scale, writing bf16 probs.
// ---------------------------------------------------------------------------
__global__ void k_softmax_bf16(const float* __restrict__ Sc,
                               unsigned short* __restrict__ P,
                               int N, float scale) {
    __shared__ float red[256];
    long row = blockIdx.x;
    const float* sp = Sc + row * N;
    float v[4];
    float mx = -1e30f;
#pragma unroll
    for (int i = 0; i < 4; ++i) {
        v[i] = sp[threadIdx.x + i * 256] * scale;
        mx = fmaxf(mx, v[i]);
    }
    red[threadIdx.x] = mx;
    __syncthreads();
    for (int wd = 128; wd > 0; wd >>= 1) {
        if (threadIdx.x < wd)
            red[threadIdx.x] = fmaxf(red[threadIdx.x], red[threadIdx.x + wd]);
        __syncthreads();
    }
    mx = red[0];
    __syncthreads();
    float sum = 0.0f;
#pragma unroll
    for (int i = 0; i < 4; ++i) {
        v[i] = __expf(v[i] - mx);
        sum += v[i];
    }
    red[threadIdx.x] = sum;
    __syncthreads();
    for (int wd = 128; wd > 0; wd >>= 1) {
        if (threadIdx.x < wd) red[threadIdx.x] += red[threadIdx.x + wd];
        __syncthreads();
    }
    float inv = 1.0f / red[0];
#pragma unroll
    for (int i = 0; i < 4; ++i)
        P[row * N + threadIdx.x + i * 256] = f2bf(v[i] * inv);
}

// ---------------------------------------------------------------------------
// V transpose: vt[(b*8+h)*64*1024 + d*1024 + k] = qkvb[(b*S+k)*1536 + 1024 + h*64 + d]
// ---------------------------------------------------------------------------
__global__ void k_transpose_v(const unsigned short* __restrict__ qkvb,
                              unsigned short* __restrict__ vt, int S) {
    long idx = (long)blockIdx.x * blockDim.x + threadIdx.x;
    if (idx >= (long)32 * 64 * 1024) return;
    int k  = (int)(idx % 1024);
    int d  = (int)((idx / 1024) % 64);
    int bh = (int)(idx / (64 * 1024));
    int b = bh >> 3, h = bh & 7;
    vt[idx] = qkvb[((long)(b * S + k)) * 1536 + 1024 + h * 64 + d];
}

// ---------------------------------------------------------------------------
// Head merge: ctxb[(b*S+q)*512 + h*64 + d] = bf16(ctx[(b*8+h)*S*64 + q*64 + d])
// ---------------------------------------------------------------------------
__global__ void k_permute_ctx(const float* __restrict__ ctx,
                              unsigned short* __restrict__ ctxb, int S) {
    long idx = (long)blockIdx.x * blockDim.x + threadIdx.x;
    if (idx >= (long)4096 * 512) return;
    int col = (int)(idx % 512);
    long row = idx / 512;
    int h = col / 64, d = col % 64;
    int b = (int)(row / S), q = (int)(row % S);
    ctxb[idx] = f2bf(ctx[((long)(b * 8 + h) * S + q) * 64 + d]);
}

// ---------------------------------------------------------------------------
// Host-side orchestration
// ---------------------------------------------------------------------------
extern "C" void kernel_launch(void* const* d_in, const int* in_sizes, int n_in,
                              void* d_out, int out_size, void* d_ws, size_t ws_size,
                              hipStream_t stream) {
    (void)in_sizes; (void)n_in; (void)out_size; (void)ws_size;

    const int*   x          = (const int*)  d_in[0];
    const float* emb        = (const float*)d_in[1];
    const float* W0         = (const float*)d_in[2];
    const float* b0         = (const float*)d_in[3];
    const float* tau0       = (const float*)d_in[4];
    const float* g0         = (const float*)d_in[5];
    const float* be0        = (const float*)d_in[6];
    const float* W1         = (const float*)d_in[7];
    const float* b1         = (const float*)d_in[8];
    const float* tau1       = (const float*)d_in[9];
    const float* g1         = (const float*)d_in[10];
    const float* be1        = (const float*)d_in[11];
    const float* attn_in_w  = (const float*)d_in[12];
    const float* attn_in_b  = (const float*)d_in[13];
    const float* attn_out_w = (const float*)d_in[14];
    const float* attn_out_b = (const float*)d_in[15];
    const float* out_w      = (const float*)d_in[16];
    const float* out_b      = (const float*)d_in[17];
    const float* log_dt     = (const float*)d_in[18];

    const int Bn = 4, S = 1024, E = 512, H = 512, V = 32000;
    const int M = Bn * S;  // 4096

    // ---- workspace carve (all 256B aligned) ----
    size_t off = 0;
    char* base = (char*)d_ws;
    auto carve = [&](size_t bytes) -> void* {
        void* p = base + off;
        off = (off + bytes + 255) & ~(size_t)255;
        return p;
    };
    unsigned short* W0b   = (unsigned short*)carve((size_t)H * E * 2);
    unsigned short* W1b   = (unsigned short*)carve((size_t)H * H * 2);
    unsigned short* Wqkvb = (unsigned short*)carve((size_t)3 * H * H * 2);
    unsigned short* Wob   = (unsigned short*)carve((size_t)H * H * 2);
    unsigned short* Wvb   = (unsigned short*)carve((size_t)V * H * 2);
    unsigned short* hb    = (unsigned short*)carve((size_t)M * H * 2);
    float*          Ibuf  = (float*)carve((size_t)M * H * 4);
    float*          Sbuf  = (float*)carve((size_t)M * H * 4);
    unsigned short* qkvb  = (unsigned short*)carve((size_t)M * 3 * H * 2);
    float*          scores= (float*)carve((size_t)32 * S * S * 4);
    unsigned short* probs = (unsigned short*)carve((size_t)32 * S * S * 2);
    unsigned short* vt    = (unsigned short*)carve((size_t)32 * 64 * S * 2);
    float*          ctx   = (float*)carve((size_t)32 * S * 64 * 4);
    unsigned short* ctxb  = (unsigned short*)carve((size_t)M * H * 2);
    unsigned short* aob   = (unsigned short*)carve((size_t)M * H * 2);

    // ---- 1. weight conversions ----
    k_f32_to_bf16<<<1024, 256, 0, stream>>>(W0,        W0b,   (long)H * E);
    k_f32_to_bf16<<<1024, 256, 0, stream>>>(W1,        W1b,   (long)H * H);
    k_f32_to_bf16<<<3072, 256, 0, stream>>>(attn_in_w, Wqkvb, (long)3 * H * H);
    k_f32_to_bf16<<<1024, 256, 0, stream>>>(attn_out_w,Wob,   (long)H * H);
    k_f32_to_bf16<<<8192, 256, 0, stream>>>(out_w,     Wvb,   (long)V * H);

    // ---- 2. embedding + positional encoding ----
    k_embed_pe<<<(M * E) / 256, 256, 0, stream>>>(x, emb, hb, S, E);

    // ---- 3. liquid layer 0 ----
    k_gemm_bf16<true><<<dim3(H / 128, M / 128, 1), 256, 0, stream>>>(
        hb, E, 0, 0, W0b, E, 0, 0, b0, Ibuf, nullptr, H, 0, 0, M, H, E, 1);
    k_liquid_scan<<<(Bn * H + 255) / 256, 256, 0, stream>>>(Ibuf, tau0, log_dt, Sbuf, Bn, S, H);
    k_tanh_ln<<<M, 256, 0, stream>>>(Sbuf, g0, be0, hb, H);

    // ---- 4. liquid layer 1 ----
    k_gemm_bf16<true><<<dim3(H / 128, M / 128, 1), 256, 0, stream>>>(
        hb, H, 0, 0, W1b, H, 0, 0, b1, Ibuf, nullptr, H, 0, 0, M, H, H, 1);
    k_liquid_scan<<<(Bn * H + 255) / 256, 256, 0, stream>>>(Ibuf, tau1, log_dt, Sbuf, Bn, S, H);
    k_tanh_ln<<<M, 256, 0, stream>>>(Sbuf, g1, be1, hb, H);

    // ---- 5. QKV projection (bf16-only output) ----
    k_gemm_bf16<true><<<dim3(3 * H / 128, M / 128, 1), 256, 0, stream>>>(
        hb, H, 0, 0, Wqkvb, H, 0, 0, attn_in_b, nullptr, qkvb, 3 * H, 0, 0,
        M, 3 * H, H, 1);

    // ---- 6. attention scores: per (b,h), A=q[1024x64], B=k[1024x64], K=64 ----
    k_gemm_bf16<true><<<dim3(S / 128, S / 128, 32), 256, 0, stream>>>(
        qkvb, 3 * H, (long)S * 3 * H, 64,
        qkvb + H, 3 * H, (long)S * 3 * H, 64,
        nullptr, scores, nullptr, S, (long)8 * S * S, (long)S * S,
        S, S, 64, 8);

    // ---- 7. softmax (scale = 1/sqrt(64)) -> bf16 probs ----
    k_softmax_bf16<<<32 * S, 256, 0, stream>>>(scores, probs, S, 0.125f);

    // ---- 8. transpose V so PV GEMM's B is K-major ----
    k_transpose_v<<<(32 * 64 * S) / 256, 256, 0, stream>>>(qkvb, vt, S);

    // ---- 9. PV: per (b,h), A=probs[1024x1024], B=vt[64x1024], C=ctx[1024x64] ----
    k_gemm_bf16<false><<<dim3(1, S / 128, 32), 256, 0, stream>>>(
        probs, S, (long)S * S, 0,
        vt, S, (long)64 * S, 0,
        nullptr, ctx, nullptr, 64, (long)S * 64, 0,
        S, 64, S, 1);

    // ---- 10. merge heads -> bf16 [4096,512] ----
    k_permute_ctx<<<(M * H) / 256, 256, 0, stream>>>(ctx, ctxb, S);

    // ---- 11. output projection of MHA (bf16-only output) ----
    k_gemm_bf16<true><<<dim3(H / 128, M / 128, 1), 256, 0, stream>>>(
        ctxb, H, 0, 0, Wob, H, 0, 0, attn_out_b, nullptr, aob, H, 0, 0,
        M, H, H, 1);

    // ---- 12. vocab projection -> d_out f32 [4096, 32000] ----
    k_gemm_bf16<true><<<dim3(V / 128, M / 128, 1), 256, 0, stream>>>(
        aob, H, 0, 0, Wvb, H, 0, 0, out_b, (float*)d_out, nullptr, V, 0, 0,
        M, V, H, 1);
}
